// QuadTreeFineEncoder_48936857371347
// MI455X (gfx1250) — matrix operand order repaired and checked by reading, not compiled, not run on hardware
//
#include <hip/hip_runtime.h>
#include <hip/hip_bf16.h>

// ---------------------------------------------------------------------------
// QuadTreeFineEncoder (bidirectional Mamba pyramid) for MI455X / gfx1250.
// GEMMs: LDS-staged bf16 tiles -> v_wmma_f32_16x16x32_bf16 (f32 accumulate).
// ---------------------------------------------------------------------------

typedef __attribute__((ext_vector_type(16))) __bf16 v16bf;
typedef __attribute__((ext_vector_type(8)))  __bf16 v8bf;
typedef __attribute__((ext_vector_type(4)))  __bf16 v4bf;
typedef __attribute__((ext_vector_type(8)))  float  v8f;

#define MB   32
#define DMOD 128
#define DI   256
#define DS   16
#define DTR  8
#define L0Q  1344
#define LTOT 2688
#define MROWS (MB * LTOT)   // 86016 = 2688 * 32, multiple of 32

enum { EPI_PLAIN = 0, EPI_SPLIT_SILU = 1, EPI_BIAS_SOFTPLUS = 2 };

__device__ __forceinline__ __bf16 f32_to_bf16(float f) {
    union { float f; unsigned u; } cv; cv.f = f;
    unsigned r = cv.u + 0x7FFFu + ((cv.u >> 16) & 1u);   // round-to-nearest-even
    unsigned short h = (unsigned short)(r >> 16);
    return __builtin_bit_cast(__bf16, h);
}

// Fast SiLU: v * sigmoid(v) with hardware v_rcp_f32 (no IEEE div sequence).
__device__ __forceinline__ float siluf(float v) {
    return v * __builtin_amdgcn_rcpf(1.0f + __expf(-v));
}

// Fast softplus: log(1+exp(v)) via v_exp/v_log; bypass for large v.
__device__ __forceinline__ float softplusf(float v) {
    return (v > 20.0f) ? v : __logf(1.0f + __expf(v));
}

// ---------------------------------------------------------------------------
// Tiled WMMA GEMM: out = A[M,K] (f32, ld=lda) x Bw[K,N] (f32, ld=ldb).
// Block = 256 threads = 8 waves; block tile = 32 (M) x 64 (N); waves 2x4.
// ALIGNED (K%64==0, N%64==0): float4 global loads, packed ds_store_b64 A tile,
//   K-chunk 64 -> two WMMAs per barrier pair, fully branch-free.
// !ALIGNED: clamp+select zero-padding (branch-free, no EXEC churn).
// Requires M % 32 == 0.
// ---------------------------------------------------------------------------
template <int MODE, bool ALIGNED>
__global__ void wmma_gemm_kernel(const float* __restrict__ A, int lda,
                                 const float* __restrict__ Bw, int ldb,
                                 float* __restrict__ out0,
                                 float* __restrict__ out1,
                                 const float* __restrict__ bias,
                                 int M, int N, int K, int ldo) {
    __shared__ __align__(16) __bf16 As[32 * 64];   // [row][k]   k-contiguous
    __shared__ __align__(16) __bf16 Bs[64 * 64];   // [col][k]   k-contiguous

    const int tid  = threadIdx.x;
    const int lane = tid & 31;
    const int wave = tid >> 5;
    const int wm   = wave >> 2;          // 0..1  (M tile within block)
    const int wn   = wave & 3;           // 0..3  (N tile within block)

    const int r0b = (int)blockIdx.y * 32;
    const int c0b = (int)blockIdx.x * 64;

    // Fragment addressing (ISA 7.12.2 layouts)
    const int aRow = wm * 16 + (lane & 15);          // A: M row within block tile
    const int kbA  = (lane < 16) ? 0 : 8;            // A K pattern {kb..kb+7, kb+16..kb+23}
    const int bCol = wn * 16 + (lane & 15);          // B/C/D: N column within block tile
    const int kbB  = (lane < 16) ? 0 : 8;            // B K pattern {kb..kb+7, kb+16..kb+23}

    const int KC = ALIGNED ? 64 : 32;

    v8f acc = {};
    for (int kk = 0; kk < K; kk += KC) {
        if constexpr (ALIGNED) {
            if (kk + KC < K)
                __builtin_prefetch(&A[(size_t)(r0b + (tid >> 3)) * lda + kk + KC], 0, 1);
            // A tile: 32x64 f32 -> bf16, float4 loads + ds_store_b64, 2/thread
#pragma unroll
            for (int j = 0; j < 2; ++j) {
                int lin = tid + j * 256;             // 512 float4s
                int row = lin >> 4, kc = (lin & 15) * 4;
                float4 av = *(const float4*)(A + (size_t)(r0b + row) * lda + kk + kc);
                v4bf pa = { f32_to_bf16(av.x), f32_to_bf16(av.y),
                            f32_to_bf16(av.z), f32_to_bf16(av.w) };
                *(v4bf*)(As + row * 64 + kc) = pa;
            }
            // B tile: 64x64 (k x col) -> transposed [col][k], float4 loads
#pragma unroll
            for (int j = 0; j < 4; ++j) {
                int lin = tid + j * 256;             // 1024 float4s
                int k = lin >> 4, colc = (lin & 15) * 4;
                float4 bv = *(const float4*)(Bw + (size_t)(kk + k) * ldb + c0b + colc);
                Bs[(colc + 0) * 64 + k] = f32_to_bf16(bv.x);
                Bs[(colc + 1) * 64 + k] = f32_to_bf16(bv.y);
                Bs[(colc + 2) * 64 + k] = f32_to_bf16(bv.z);
                Bs[(colc + 3) * 64 + k] = f32_to_bf16(bv.w);
            }
        } else {
            // Branch-free zero-padding: clamp index in-bounds, select 0.
#pragma unroll
            for (int j = 0; j < 4; ++j) {
                int lin = tid + j * 256;             // 1024 elems
                int row = lin >> 5, k = lin & 31;
                int kc = (kk + k < K) ? (kk + k) : (K - 1);
                float v = A[(size_t)(r0b + row) * lda + kc];
                As[row * 64 + k] = f32_to_bf16((kk + k < K) ? v : 0.0f);
            }
#pragma unroll
            for (int j = 0; j < 8; ++j) {
                int lin = tid + j * 256;             // 2048 elems
                int k = lin >> 6, col = lin & 63;
                bool ok = (kk + k < K) && (c0b + col < N);
                int kc = (kk + k < K) ? (kk + k) : (K - 1);
                int cc = (c0b + col < N) ? (c0b + col) : (N - 1);
                float v = Bw[(size_t)kc * ldb + cc];
                Bs[col * 64 + k] = f32_to_bf16(ok ? v : 0.0f);
            }
        }
        __syncthreads();

        // ---- branch-free fragment loads from LDS + WMMA ----
#pragma unroll
        for (int hc = 0; hc < (ALIGNED ? 2 : 1); ++hc) {
            const __bf16* ap = As + aRow * 64 + hc * 32;
            const __bf16* bp = Bs + bCol * 64 + hc * 32;
            v8bf alo = *(const v8bf*)(ap + kbA);
            v8bf ahi = *(const v8bf*)(ap + kbA + 16);
            v8bf blo = *(const v8bf*)(bp + kbB);
            v8bf bhi = *(const v8bf*)(bp + kbB + 16);
            v16bf a, b;
#pragma unroll
            for (int e = 0; e < 8; ++e) {
                a[e] = alo[e];  a[8 + e] = ahi[e];
                b[e] = blo[e];  b[8 + e] = bhi[e];
            }
            acc = __builtin_amdgcn_wmma_f32_16x16x32_bf16(
                /*neg_a=*/false, a, /*neg_b=*/false, b,
                /*c_mod=*/(short)0, acc, /*reuse_a=*/false, /*reuse_b=*/false);
        }
        __syncthreads();
    }

    // ---- epilogue (compile-time mode; aligned kernels never mask columns) ----
    const int col   = c0b + bCol;
    const int rbase = r0b + wm * 16 + ((lane < 16) ? 0 : 8);
#pragma unroll
    for (int v = 0; v < 8; ++v) {
        int r = rbase + v;
        if constexpr (!ALIGNED) { if (col >= N) continue; }
        float val = acc[v];
        if constexpr (MODE == EPI_PLAIN) {
            out0[(size_t)r * ldo + col] = val;
        } else if constexpr (MODE == EPI_SPLIT_SILU) {
            int h = N >> 1;
            if (col < h) out0[(size_t)r * h + col] = val;
            else         out1[(size_t)r * h + (col - h)] = siluf(val);
        } else { // EPI_BIAS_SOFTPLUS
            out0[(size_t)r * ldo + col] = softplusf(val + bias[col]);
        }
    }
}

// ---------------------------------------------------------------------------
// x = concat(feat0, feat1) along the sequence axis.
// ---------------------------------------------------------------------------
__global__ void init_x_kernel(const float* __restrict__ f0,
                              const float* __restrict__ f1,
                              float* __restrict__ x) {
    int bt = blockIdx.x;                 // 0 .. MB*LTOT-1
    int b = bt / LTOT, t = bt % LTOT;
    int c = threadIdx.x;                 // 0 .. 127
    float v = (t < L0Q) ? f0[((size_t)b * L0Q + t) * DMOD + c]
                        : f1[((size_t)b * L0Q + (t - L0Q)) * DMOD + c];
    x[(size_t)bt * DMOD + c] = v;
}

__global__ void reverse_kernel(const float* __restrict__ x, float* __restrict__ xb) {
    int bt = blockIdx.x;
    int b = bt / LTOT, t = bt % LTOT;
    int c = threadIdx.x;
    xb[(size_t)bt * DMOD + c] = x[((size_t)b * LTOT + (LTOT - 1 - t)) * DMOD + c];
}

// ---------------------------------------------------------------------------
// Causal depthwise conv (k=4) + bias + SiLU.  Branch-free boundary handling.
// ---------------------------------------------------------------------------
__global__ void conv_silu_kernel(const float* __restrict__ xi, float* __restrict__ xc,
                                 const float* __restrict__ cw, const float* __restrict__ cb) {
    int bt = blockIdx.x;
    int b = bt / LTOT, t = bt % LTOT;
    int d = threadIdx.x;                 // 0 .. 255
    float acc = cb[d];
#pragma unroll
    for (int j = 0; j < 4; ++j) {
        int tt = t - 3 + j;
        int tc = (tt >= 0) ? tt : 0;                       // clamp in-bounds
        float xv = xi[((size_t)b * LTOT + tc) * DI + d];
        acc += ((tt >= 0) ? xv : 0.0f) * cw[d * 4 + j];    // select, no EXEC churn
    }
    xc[(size_t)bt * DI + d] = siluf(acc);
}

// ---------------------------------------------------------------------------
// Selective scan: one lane per (batch, channel); 16-state recurrence in regs.
// B_t / C_t (32 floats) broadcast via LDS each step. Fuses y*silu(z) -> xc.
// ---------------------------------------------------------------------------
__global__ void scan_kernel(float* __restrict__ xc,          // u in, y*silu(z) out
                            const float* __restrict__ dtb,
                            const float* __restrict__ xdbl,  // [., 40]: B at 8, C at 24
                            const float* __restrict__ sz,
                            const float* __restrict__ A_log, // [DI, DS]
                            const float* __restrict__ Dp) {  // [DI]
    const int b = blockIdx.x;            // batch
    const int d = threadIdx.x;           // channel 0..255
    __shared__ float sBC[32];

    float A[DS];
#pragma unroll
    for (int n = 0; n < DS; ++n) A[n] = -__expf(A_log[d * DS + n]);
    const float Dpar = Dp[d];

    float h[DS];
#pragma unroll
    for (int n = 0; n < DS; ++n) h[n] = 0.0f;

    for (int t = 0; t < LTOT; ++t) {
        const size_t rowbase = ((size_t)b * LTOT + t);
        if (d < 32) sBC[d] = xdbl[rowbase * 40 + 8 + d];
        __syncthreads();
        const size_t idx = rowbase * DI + d;
        const float dtv = dtb[idx];
        const float u = xc[idx];
        const float du = dtv * u;
        float y = 0.0f;
#pragma unroll
        for (int n = 0; n < DS; ++n) {
            float dA = __expf(dtv * A[n]);
            h[n] = dA * h[n] + du * sBC[n];
            y += h[n] * sBC[16 + n];
        }
        y += u * Dpar;
        xc[idx] = y * sz[idx];
        __syncthreads();
    }
}

// ---------------------------------------------------------------------------
// dx = 0.5*(dxf + reverse(dxb)); layernorm over 128 channels; x += result.
// Wave32 shuffle butterflies + 4-partial LDS combine.
// ---------------------------------------------------------------------------
__device__ __forceinline__ float block128_sum(float v, float* part,
                                              int lane, int wave, int c) {
#pragma unroll
    for (int m = 16; m > 0; m >>= 1) v += __shfl_xor(v, m, 32);
    if (lane == 0) part[wave] = v;
    __syncthreads();
    float s = part[0] + part[1] + part[2] + part[3];
    __syncthreads();
    return s;
}

__global__ void combine_ln_kernel(float* __restrict__ x,
                                  const float* __restrict__ dxf,
                                  const float* __restrict__ dxb,
                                  const float* __restrict__ lnw,
                                  const float* __restrict__ lnb) {
    int bt = blockIdx.x;
    int b = bt / LTOT, t = bt % LTOT;
    int c = threadIdx.x;                 // 0 .. 127
    int lane = c & 31, wave = c >> 5;
    __shared__ float part[4];

    size_t iF = (size_t)bt * DMOD + c;
    size_t iB = ((size_t)b * LTOT + (LTOT - 1 - t)) * DMOD + c;
    float dx = 0.5f * (dxf[iF] + dxb[iB]);

    float mu = block128_sum(dx, part, lane, wave, c) * (1.0f / DMOD);
    float dv = dx - mu;
    float var = block128_sum(dv * dv, part, lane, wave, c) * (1.0f / DMOD);
    float nrm = dv * rsqrtf(var + 1e-5f);
    x[iF] = nrm * lnw[c] + lnb[c] + x[iF];
}

// ---------------------------------------------------------------------------
// Pyramid: out[half][m][c][y][x] = seg2[y,x] + seg1[y/2,x/2] + seg0[y/4,x/4].
// ---------------------------------------------------------------------------
__global__ void pyramid_kernel(const float* __restrict__ x, float* __restrict__ out) {
    size_t gid = (size_t)blockIdx.x * blockDim.x + threadIdx.x; // 2*32*128*32*32
    int xx = (int)(gid & 31);
    int yy = (int)((gid >> 5) & 31);
    int c  = (int)((gid >> 10) & 127);
    int m  = (int)((gid >> 17) & 31);
    int half = (int)(gid >> 22);
    size_t base = (size_t)m * LTOT + (size_t)half * L0Q;
    int i2 = 320 + yy * 32 + xx;
    int i1 = 64 + (yy >> 1) * 16 + (xx >> 1);
    int i0 = (yy >> 2) * 8 + (xx >> 2);
    out[gid] = x[(base + i2) * DMOD + c] +
               x[(base + i1) * DMOD + c] +
               x[(base + i0) * DMOD + c];
}

// ---------------------------------------------------------------------------
// Host-side orchestration.
// ---------------------------------------------------------------------------
extern "C" void kernel_launch(void* const* d_in, const int* in_sizes, int n_in,
                              void* d_out, int out_size, void* d_ws, size_t ws_size,
                              hipStream_t stream) {
    const float* f0     = (const float*)d_in[0];
    const float* f1     = (const float*)d_in[1];
    const float* W_in   = (const float*)d_in[2];   // (2,2,128,512)
    const float* conv_w = (const float*)d_in[3];   // (2,2,256,4)
    const float* conv_b = (const float*)d_in[4];   // (2,2,256)
    const float* W_x    = (const float*)d_in[5];   // (2,2,256,40)
    const float* W_dt   = (const float*)d_in[6];   // (2,2,8,256)
    const float* b_dt   = (const float*)d_in[7];   // (2,2,256)
    const float* A_log  = (const float*)d_in[8];   // (2,2,256,16)
    const float* D_par  = (const float*)d_in[9];   // (2,2,256)
    const float* W_out  = (const float*)d_in[10];  // (2,2,256,128)
    const float* ln_w   = (const float*)d_in[11];  // (2,128)
    const float* ln_b   = (const float*)d_in[12];  // (2,128)

    const size_t M = MROWS;
    // Workspace layout (floats). xi buffer is reused as dt after the conv.
    float* p    = (float*)d_ws;
    float* x    = p; p += M * (size_t)DMOD;
    float* xb   = p; p += M * (size_t)DMOD;
    float* xi   = p; p += M * (size_t)DI;      // also dt
    float* sz   = p; p += M * (size_t)DI;
    float* xc   = p; p += M * (size_t)DI;
    float* xdbl = p; p += M * (size_t)40;
    float* dxf  = p; p += M * (size_t)DMOD;
    float* dxb  = p; p += M * (size_t)DMOD;
    size_t needed = (size_t)(p - (float*)d_ws) * sizeof(float);
    if (ws_size < needed) return;              // insufficient scratch: bail safely

    const int mBlk = MROWS / 32;               // 2688 block-rows

    init_x_kernel<<<MB * LTOT, DMOD, 0, stream>>>(f0, f1, x);

    for (int i = 0; i < 2; ++i) {
        reverse_kernel<<<MB * LTOT, DMOD, 0, stream>>>(x, xb);
        for (int dir = 0; dir < 2; ++dir) {
            const int ld = i * 2 + dir;
            const float* src = dir ? xb : x;
            float* dxd = dir ? dxb : dxf;

            // xz = src @ W_in -> xi (cols 0..255), silu(z) -> sz   (N=512, K=128)
            wmma_gemm_kernel<EPI_SPLIT_SILU, true><<<dim3(8, mBlk), 256, 0, stream>>>(
                src, DMOD, W_in + (size_t)ld * 128 * 512, 512,
                xi, sz, nullptr, MROWS, 512, 128, 0);

            // causal depthwise conv + SiLU
            conv_silu_kernel<<<MB * LTOT, DI, 0, stream>>>(
                xi, xc, conv_w + (size_t)ld * DI * 4, conv_b + (size_t)ld * DI);

            // xdbl = xc @ W_x  (N=40, zero-padded tile)
            wmma_gemm_kernel<EPI_PLAIN, false><<<dim3(1, mBlk), 256, 0, stream>>>(
                xc, DI, W_x + (size_t)ld * DI * 40, 40,
                xdbl, nullptr, nullptr, MROWS, 40, DI, 40);

            // dt = softplus(xdbl[:, :8] @ W_dt + b_dt)  (K=8 zero-padded to 32)
            wmma_gemm_kernel<EPI_BIAS_SOFTPLUS, false><<<dim3(4, mBlk), 256, 0, stream>>>(
                xdbl, 40, W_dt + (size_t)ld * DTR * DI, DI,
                xi, nullptr, b_dt + (size_t)ld * DI,
                MROWS, DI, DTR, DI);

            // selective scan (fuses y * silu(z) back into xc)
            scan_kernel<<<MB, DI, 0, stream>>>(
                xc, xi, xdbl, sz,
                A_log + (size_t)ld * DI * DS, D_par + (size_t)ld * DI);

            // dx_dir = xc @ W_out   (N=128, K=256)
            wmma_gemm_kernel<EPI_PLAIN, true><<<dim3(2, mBlk), 256, 0, stream>>>(
                xc, DI, W_out + (size_t)ld * DI * DMOD, DMOD,
                dxd, nullptr, nullptr, MROWS, DMOD, DI, DMOD);
        }
        combine_ln_kernel<<<MB * LTOT, DMOD, 0, stream>>>(
            x, dxf, dxb, ln_w + (size_t)i * DMOD, ln_b + (size_t)i * DMOD);
    }

    pyramid_kernel<<<(2u * MB * DMOD * 32 * 32) / 256, 256, 0, stream>>>(x, (float*)d_out);
}
